// MultiHeadAttention_58952721105411
// MI455X (gfx1250) — compile-verified, matrix operand order checked by reading, not run on hardware
//
#include <hip/hip_runtime.h>
#include <hip/hip_bf16.h>

// ---------------------------------------------------------------------------
// MHA for MI455X (gfx1250, wave32, WMMA).
//  - fused attention: 1 GiB attention_weights written exactly once, never
//    re-read from HBM; scores/probs staged in LDS as f16 (64 KB per 16-row
//    q-tile -> 4-wave workgroups, 256 KB of the WGP's 320 KB LDS, one wave
//    per SIMD32).
//  - all matmuls on v_wmma_f32_16x16x32_f16; GEMMs use 64x64 per-wave tiles
//    (16 WMMA per K-step, ~32 FLOP/B from L2).
// ---------------------------------------------------------------------------

typedef _Float16 half8_t __attribute__((ext_vector_type(8)));
typedef _Float16 v16h    __attribute__((ext_vector_type(16)));
typedef float    v8f     __attribute__((ext_vector_type(8)));

constexpr int Bn = 4, Sn = 2048, En = 1024, Hn = 16, Dn = 64;

#define DEVFN __device__ __forceinline__

// Load one 16x32 f16 WMMA A/B fragment from a row-major matrix (global or LDS).
// p points at element (row_of_this_lane, kt + (lane>=16 ? 8 : 0)).
// ISA layout: lanes 0-15 hold K = {0..7, 16..23}, lanes 16-31 hold {8..15, 24..31}.
DEVFN v16h load_frag16(const _Float16* p) {
  half8_t lo = *(const half8_t*)(p);
  half8_t hi = *(const half8_t*)(p + 16);
  v16h f;
#pragma unroll
  for (int i = 0; i < 8; ++i) { f[i] = lo[i]; f[i + 8] = hi[i]; }
  return f;
}

DEVFN v8f wmma_f16(v16h a, v16h b, v8f c) {
  // 8 args: (neg_a, A, neg_b, B, c_mod, C, reuse_a, reuse_b)
  return __builtin_amdgcn_wmma_f32_16x16x32_f16(false, a, false, b, (short)0, c,
                                                false, false);
}

// ---------------------------------------------------------------------------
// fp32 -> fp16 conversion
// ---------------------------------------------------------------------------
__global__ __launch_bounds__(256) void cvt_f32_to_f16(const float* __restrict__ in,
                                                      _Float16* __restrict__ out,
                                                      int n) {
  int i = blockIdx.x * blockDim.x + threadIdx.x;
  if (i < n) out[i] = (_Float16)in[i];
}

// V [B,S,H,D] (f16) -> Vt [B,H,D,S] (f16) so P·V B-fragments are contiguous.
__global__ __launch_bounds__(256) void transpose_v(const _Float16* __restrict__ v,
                                                   _Float16* __restrict__ vt) {
  int i = blockIdx.x * blockDim.x + threadIdx.x;
  if (i >= Bn * Sn * En) return;
  int d = i % Dn;
  int h = (i / Dn) % Hn;
  int s = (i / En) % Sn;
  int b = i / (Sn * En);
  vt[((size_t)(b * Hn + h) * Dn + d) * Sn + s] = v[i];
}

// ---------------------------------------------------------------------------
// C[M,N] = A[M,K] * W[N,K]^T + bias[N]
// One wave owns a 64x64 C tile: 4 A-frags x 4 B-frags -> 16 WMMA per K-step.
// ---------------------------------------------------------------------------
template <bool F16OUT>
__global__ __launch_bounds__(256) void gemm_bias_wmma(
    const _Float16* __restrict__ A,  // [M,K] row-major
    const _Float16* __restrict__ W,  // [N,K] row-major
    const float* __restrict__ bias,  // [N]
    void* __restrict__ C,            // [M,N]
    int M, int N, int K) {
  const int wave = blockIdx.x * (blockDim.x >> 5) + (threadIdx.x >> 5);
  const int lane = threadIdx.x & 31;
  const int mTiles = M >> 6;
  const int tm = wave % mTiles;
  const int tn = wave / mTiles;
  if (tn >= (N >> 6)) return;
  const int m0 = tm << 6, n0 = tn << 6;
  const int nl = lane & 15, hi = lane >> 4;

  v8f acc[4][4] = {};
  const _Float16* aBase = A + (size_t)(m0 + nl) * K + (hi << 3);
  const _Float16* bBase = W + (size_t)(n0 + nl) * K + (hi << 3);

  for (int kt = 0; kt < K; kt += 32) {
    __builtin_prefetch(aBase + kt + 64, 0, 1);  // global_prefetch
    v16h a[4];
#pragma unroll
    for (int i = 0; i < 4; ++i)
      a[i] = load_frag16(aBase + (size_t)(i * 16) * K + kt);
#pragma unroll
    for (int j = 0; j < 4; ++j) {
      v16h b = load_frag16(bBase + (size_t)(j * 16) * K + kt);
#pragma unroll
      for (int i = 0; i < 4; ++i)
        acc[i][j] = wmma_f16(a[i], b, acc[i][j]);
    }
  }
  // C layout: VGPR r of lane L holds element (M = r + 8*(L>=16), N = L&15)
#pragma unroll
  for (int j = 0; j < 4; ++j) {
    const int n = n0 + j * 16 + nl;
    const float bv = bias[n];
#pragma unroll
    for (int i = 0; i < 4; ++i) {
#pragma unroll
      for (int r = 0; r < 8; ++r) {
        const size_t idx = (size_t)(m0 + i * 16 + r + 8 * hi) * N + n;
        const float v = acc[i][j][r] + bv;
        if (F16OUT) ((_Float16*)C)[idx] = (_Float16)v;
        else        ((float*)C)[idx]    = v;
      }
    }
  }
}

// ---------------------------------------------------------------------------
// Fused attention: 4 waves per workgroup; each wave owns one 16-row q tile
// and a private 64 KB f16 score slice in LDS (4 x 64 KB = 256 KB <= 320 KB).
//   scores = Q K^T / 8        -> f16 LDS slice
//   softmax over S            -> fp32 attention_weights out (coalesced),
//                                normalized probs kept in LDS as f16
//   out = P V                 -> WMMA, A-frags loaded straight from LDS
//   softmax over D            -> f16 attention output for final projection
// ---------------------------------------------------------------------------
__global__ __launch_bounds__(128) void attn_fused(
    const _Float16* __restrict__ Q,   // [B,S,E] (= [B,S,H,D])
    const _Float16* __restrict__ Km,  // [B,S,E]
    const _Float16* __restrict__ Vt,  // [B,H,D,S]
    float* __restrict__ aw,           // [B,H,S,S]
    _Float16* __restrict__ O16) {     // [B,S,E]
  extern __shared__ _Float16 smem[];  // [4][16][Sn] f16
  const int lane = threadIdx.x & 31;
  const int wv   = threadIdx.x >> 5;           // wave in block: 0..3
  const int bid  = blockIdx.x;                 // over B*H*(S/64)
  const int qb = bid & 31;                     // 64-row q block (S/64 = 32)
  const int h  = (bid >> 5) & 15;
  const int b  = bid >> 9;
  const int q0 = qb * 64 + wv * 16;            // this wave's 16-row q tile
  const int nl = lane & 15, hi = lane >> 4;

  _Float16* sc = smem + (size_t)wv * 16 * Sn;  // this wave's score slice

  // Q tile fragments (K dim = D = 64 -> two fragments)
  const _Float16* qbase =
      Q + (size_t)(b * Sn + q0 + nl) * En + h * Dn + (hi << 3);
  const v16h aq0 = load_frag16(qbase);
  const v16h aq1 = load_frag16(qbase + 32);

  // ---- scores = Q K^T * (1/sqrt(D)) --------------------------------------
  for (int kt = 0; kt < Sn; kt += 16) {
    const _Float16* kb =
        Km + (size_t)(b * Sn + kt + nl) * En + h * Dn + (hi << 3);
    v8f c = {};
    c = wmma_f16(aq0, load_frag16(kb), c);
    c = wmma_f16(aq1, load_frag16(kb + 32), c);
#pragma unroll
    for (int r = 0; r < 8; ++r)
      sc[(r + 8 * hi) * Sn + kt + nl] = (_Float16)(c[r] * 0.125f);
  }
  __syncthreads();

  // ---- softmax over S, emit fp32 attention_weights -----------------------
  for (int m = 0; m < 16; ++m) {
    _Float16* row = sc + m * Sn;
    float mx = -INFINITY;
    for (int j = lane; j < Sn; j += 32) mx = fmaxf(mx, (float)row[j]);
#pragma unroll
    for (int off = 16; off > 0; off >>= 1) mx = fmaxf(mx, __shfl_xor(mx, off, 32));
    float sum = 0.f;
    for (int j = lane; j < Sn; j += 32) sum += __expf((float)row[j] - mx);
#pragma unroll
    for (int off = 16; off > 0; off >>= 1) sum += __shfl_xor(sum, off, 32);
    const float inv = 1.f / sum;
    float* awrow = aw + ((size_t)(b * Hn + h) * Sn + q0 + m) * Sn;
    for (int j = lane; j < Sn; j += 32) {
      float p = __expf((float)row[j] - mx) * inv;
      row[j] = (_Float16)p;  // prob kept in LDS for P·V
      awrow[j] = p;          // coalesced 1 GiB write, written exactly once
    }
  }
  __syncthreads();

  // ---- out = P V (A-fragments straight from LDS as packed f16) -----------
  v8f acc[4] = {};
  for (int kt = 0; kt < Sn; kt += 32) {
    v16h ap = load_frag16(sc + nl * Sn + kt + (hi << 3));  // ds_load_b128 x2
#pragma unroll
    for (int t = 0; t < 4; ++t) {
      const _Float16* vb =
          Vt + (size_t)((b * Hn + h) * Dn + t * 16 + nl) * Sn + kt + (hi << 3);
      acc[t] = wmma_f16(ap, load_frag16(vb), acc[t]);
    }
  }
  __syncthreads();

  // ---- softmax over D on the 16x64 tile, write f16 for final projection --
  _Float16* ot = sc;  // reuse front 2 KB of this wave's slice
#pragma unroll
  for (int t = 0; t < 4; ++t)
#pragma unroll
    for (int r = 0; r < 8; ++r)
      ot[(r + 8 * hi) * 64 + t * 16 + nl] = (_Float16)acc[t][r];
  __syncthreads();
  if (lane < 16) {
    const int m = lane;
    float mx = -INFINITY;
    for (int d = 0; d < 64; ++d) mx = fmaxf(mx, (float)ot[m * 64 + d]);
    float sum = 0.f;
    for (int d = 0; d < 64; ++d) sum += __expf((float)ot[m * 64 + d] - mx);
    const float inv = 1.f / sum;
    _Float16* orow = O16 + (size_t)(b * Sn + q0 + m) * En + h * Dn;
    for (int d = 0; d < 64; ++d)
      orow[d] = (_Float16)(__expf((float)ot[m * 64 + d] - mx) * inv);
  }
}

// ---------------------------------------------------------------------------
// Launcher
// ---------------------------------------------------------------------------
extern "C" void kernel_launch(void* const* d_in, const int* in_sizes, int n_in,
                              void* d_out, int out_size, void* d_ws, size_t ws_size,
                              hipStream_t stream) {
  const float* key   = (const float*)d_in[0];
  const float* query = (const float*)d_in[1];
  const float* value = (const float*)d_in[2];
  const float* wk_w  = (const float*)d_in[3];
  const float* wk_b  = (const float*)d_in[4];
  const float* wq_w  = (const float*)d_in[5];
  const float* wq_b  = (const float*)d_in[6];
  const float* wv_w  = (const float*)d_in[7];
  const float* wv_b  = (const float*)d_in[8];
  const float* out_w = (const float*)d_in[9];
  const float* out_b = (const float*)d_in[10];

  float* outp = (float*)d_out;                // [B,S,E]
  float* aw   = outp + (size_t)Bn * Sn * En;  // [B,H,S,S]

  const size_t ACT = (size_t)Bn * Sn * En;  // 8,388,608
  const size_t WEI = (size_t)En * En;       // 1,048,576

  size_t off = 0;
  auto alloc16 = [&](size_t elems) -> _Float16* {
    off = (off + 255) & ~(size_t)255;
    _Float16* p = (_Float16*)((char*)d_ws + off);
    off += elems * sizeof(_Float16);
    return p;
  };
  _Float16* qi = alloc16(ACT);
  _Float16* ki = alloc16(ACT);
  _Float16* vi = alloc16(ACT);
  _Float16* wq = alloc16(WEI);
  _Float16* wk = alloc16(WEI);
  _Float16* wv = alloc16(WEI);
  _Float16* wo = alloc16(WEI);
  _Float16* Qp = alloc16(ACT);  // projected Q [B,S,E]
  _Float16* Kp = alloc16(ACT);
  _Float16* Vp = alloc16(ACT);
  _Float16* Vt = alloc16(ACT);  // V transposed [B,H,D,S]
  _Float16* AO = alloc16(ACT);  // attention output (post feature-softmax)

  const int M = Bn * Sn, N = En, K = En;
  const int cvtBlocksA = (int)((ACT + 255) / 256);
  const int cvtBlocksW = (int)((WEI + 255) / 256);
  const int gemmBlocks = (M / 64) * (N / 64) / 8;  // 64x64/wave, 8 waves/block

  // fp32 -> fp16
  cvt_f32_to_f16<<<cvtBlocksA, 256, 0, stream>>>(query, qi, (int)ACT);
  cvt_f32_to_f16<<<cvtBlocksA, 256, 0, stream>>>(key,   ki, (int)ACT);
  cvt_f32_to_f16<<<cvtBlocksA, 256, 0, stream>>>(value, vi, (int)ACT);
  cvt_f32_to_f16<<<cvtBlocksW, 256, 0, stream>>>(wq_w, wq, (int)WEI);
  cvt_f32_to_f16<<<cvtBlocksW, 256, 0, stream>>>(wk_w, wk, (int)WEI);
  cvt_f32_to_f16<<<cvtBlocksW, 256, 0, stream>>>(wv_w, wv, (int)WEI);
  cvt_f32_to_f16<<<cvtBlocksW, 256, 0, stream>>>(out_w, wo, (int)WEI);

  // Q/K/V projections (WMMA)
  gemm_bias_wmma<true><<<gemmBlocks, 256, 0, stream>>>(qi, wq, wq_b, Qp, M, N, K);
  gemm_bias_wmma<true><<<gemmBlocks, 256, 0, stream>>>(ki, wk, wk_b, Kp, M, N, K);
  gemm_bias_wmma<true><<<gemmBlocks, 256, 0, stream>>>(vi, wv, wv_b, Vp, M, N, K);

  // V -> [B,H,D,S]
  transpose_v<<<cvtBlocksA, 256, 0, stream>>>(Vp, Vt);

  // fused attention: 4 waves/WG, 256 KB dynamic LDS (CDNA5 WGP has 320 KB)
  const int attnBlocks = Bn * Hn * (Sn / 64);  // 2048
  const int attnLds = 4 * 16 * Sn * (int)sizeof(_Float16);  // 262144 B
  attn_fused<<<attnBlocks, 128, attnLds, stream>>>(Qp, Kp, Vt, aw, AO);

  // final projection (WMMA, fp32 out + bias)
  gemm_bias_wmma<false><<<gemmBlocks, 256, 0, stream>>>(AO, wo, out_b, outp, M, N, K);
}
